// QLoRABigNet_15977278341791
// MI455X (gfx1250) — matrix operand order repaired and checked by reading, not compiled, not run on hardware
//
#include <hip/hip_runtime.h>
#include <hip/hip_bf16.h>
#include <stdint.h>

typedef _Float16 half_t;
typedef _Float16 v16h __attribute__((ext_vector_type(16)));
typedef _Float16 v8h  __attribute__((ext_vector_type(8)));
typedef float    v8f  __attribute__((ext_vector_type(8)));

#define N_TOK 16384
#define DIM   1024
#define RANK  32
#define NGRP  (DIM / 16)

#define BM  256  // rows per block (main GEMM)
#define BN  128  // cols per block (main GEMM)
#define BKS 32   // K staged in LDS per pipeline stage (1 WMMA K-step)
#define NSTG (DIM / BKS)
#define KP  40   // padded row length in halves (80B stride: 16B aligned, conflict-free)
#define BML 128  // rows per block (lora GEMM)

// ---- fragment builders per CDNA5 wave32 WMMA VGPR layouts (CDNA5 ISA) ----
__device__ __forceinline__ v16h load_a_frag(const half_t* __restrict__ row_ptr, int laneHi) {
    const v8h lo = *(const v8h*)(row_ptr + 8 * laneHi);
    const v8h hi = *(const v8h*)(row_ptr + 16 + 8 * laneHi);
    v16h a;
#pragma unroll
    for (int i = 0; i < 8; ++i) { a[i] = lo[i]; a[8 + i] = hi[i]; }
    return a;
}
__device__ __forceinline__ v16h load_b_frag(const half_t* __restrict__ col_ptr, int laneHi) {
    const v8h lo = *(const v8h*)(col_ptr + 8 * laneHi);
    const v8h hi = *(const v8h*)(col_ptr + 16 + 8 * laneHi);
    v16h b;
#pragma unroll
    for (int i = 0; i < 8; ++i) { b[i] = lo[i]; b[8 + i] = hi[i]; }
    return b;
}

// async global->LDS 128b copy (ASYNCcnt-tracked, no VGPR staging)
__device__ __forceinline__ void async_copy_b128(const void* gptr, void* lds_ptr) {
    const uint32_t loff = (uint32_t)(size_t)lds_ptr;        // LDS aperture: addr[31:0]
    const uint64_t ga   = (uint64_t)(size_t)gptr;
    asm volatile("global_load_async_to_lds_b128 %0, %1, off"
                 :: "v"(loff), "v"(ga) : "memory");
}
// wait for all but the newest 6 async loads (6 = per-wave loads of one stage);
// async loads complete in order, so this releases exactly the previous stage.
__device__ __forceinline__ void wait_async_prev() {
    asm volatile("s_wait_asynccnt 6" ::: "memory");
}
__device__ __forceinline__ void wait_async_all() {
    asm volatile("s_wait_asynccnt 0" ::: "memory");
}

// =====================  per-layer weight dequant: q(int32 4-bit) -> f16  =====================
__global__ __launch_bounds__(256) void dequant_kernel(
    const int* __restrict__ Q, const float* __restrict__ S, half_t* __restrict__ W)
{
    const int idx = blockIdx.x * 256 + threadIdx.x;   // one 8-element chunk each
    const int o = idx >> 7;
    const int k = (idx & 127) * 8;                    // stays within one scale group of 16
    const int4 q0 = *(const int4*)&Q[o * DIM + k];
    const int4 q1 = *(const int4*)&Q[o * DIM + k + 4];
    const float s = S[o * NGRP + (k >> 4)];
    v8h w;
    w[0] = (half_t)(((float)q0.x * (2.0f / 15.0f) - 1.0f) * s);
    w[1] = (half_t)(((float)q0.y * (2.0f / 15.0f) - 1.0f) * s);
    w[2] = (half_t)(((float)q0.z * (2.0f / 15.0f) - 1.0f) * s);
    w[3] = (half_t)(((float)q0.w * (2.0f / 15.0f) - 1.0f) * s);
    w[4] = (half_t)(((float)q1.x * (2.0f / 15.0f) - 1.0f) * s);
    w[5] = (half_t)(((float)q1.y * (2.0f / 15.0f) - 1.0f) * s);
    w[6] = (half_t)(((float)q1.z * (2.0f / 15.0f) - 1.0f) * s);
    w[7] = (half_t)(((float)q1.w * (2.0f / 15.0f) - 1.0f) * s);
    *(v8h*)&W[o * DIM + k] = w;
}

// =====================  main QLoRA GEMM (64x64 wave tile, double-buffered async DMA)  =====================
// out[n,o] = sum_k x[n,k]*W[o,k] + bias[o] + sum_r t[n,r]*B16[o,r]  (+res) (relu)
template <int RELU, int F16OUT, int ADDRES>
__global__ __launch_bounds__(256) void qlora_gemm_kernel(
    const half_t* __restrict__ X,    // [N_TOK, DIM] f16
    const half_t* __restrict__ W,    // [DIM, DIM] f16 (dequantized)
    const float*  __restrict__ Bias, // [DIM]
    const half_t* __restrict__ T,    // [N_TOK, RANK] f16
    const half_t* __restrict__ B16,  // [DIM, RANK] f16
    const float*  __restrict__ Res,  // [N_TOK, DIM] (ADDRES only)
    half_t* __restrict__ O16,        // (F16OUT)
    float*  __restrict__ O32)        // (!F16OUT)
{
    __shared__ __align__(16) half_t lds_x[2][BM][KP];
    __shared__ __align__(16) half_t lds_w[2][BN][KP];

    const int tid    = threadIdx.x;
    const int lane   = tid & 31;
    const int wid    = tid >> 5;
    const int laneHi = lane >> 4;
    const int laneLo = lane & 15;
    const int n0 = blockIdx.y * BM;
    const int o0 = blockIdx.x * BN;
    const int wm = wid >> 1;  // 0..3 -> 64 rows each
    const int wn = wid & 1;   // 0..1 -> 64 cols each

    // per-thread staging coordinates (6 async b128 per stage: 4 for x, 2 for w)
    const int srow = tid >> 2;            // 0..63 base row group
    const int skk  = (tid & 3) * 8;       // 0,8,16,24

    v8f c[4][4];
#pragma unroll
    for (int i = 0; i < 4; ++i)
#pragma unroll
        for (int j = 0; j < 4; ++j) c[i][j] = v8f{};

    // issue stage 0 DMA
#pragma unroll
    for (int i = 0; i < 4; ++i)
        async_copy_b128(&X[(size_t)(n0 + srow + 64 * i) * DIM + skk], &lds_x[0][srow + 64 * i][skk]);
#pragma unroll
    for (int i = 0; i < 2; ++i)
        async_copy_b128(&W[(size_t)(o0 + srow + 64 * i) * DIM + skk], &lds_w[0][srow + 64 * i][skk]);

#pragma unroll 2
    for (int s = 0; s < NSTG; ++s) {
        const int cur = s & 1;
        const int k0 = s * BKS;
        if (s + 1 < NSTG) {   // issue next stage DMA into the other buffer
            const int nxt = (s + 1) & 1;
            const int kn = k0 + BKS;
#pragma unroll
            for (int i = 0; i < 4; ++i)
                async_copy_b128(&X[(size_t)(n0 + srow + 64 * i) * DIM + kn + skk],
                                &lds_x[nxt][srow + 64 * i][skk]);
#pragma unroll
            for (int i = 0; i < 2; ++i)
                async_copy_b128(&W[(size_t)(o0 + srow + 64 * i) * DIM + kn + skk],
                                &lds_w[nxt][srow + 64 * i][skk]);
            if (s + 2 < NSTG) {  // L2 prefetch for the stage after next
                __builtin_prefetch(&X[(size_t)(n0 + srow) * DIM + kn + BKS + skk], 0, 0);
                __builtin_prefetch(&W[(size_t)(o0 + srow) * DIM + kn + BKS + skk], 0, 0);
            }
            wait_async_prev();   // oldest 6 (stage s) have landed
        } else {
            wait_async_all();
        }
        __syncthreads();

        v16h bfr[4];
#pragma unroll
        for (int tn = 0; tn < 4; ++tn)
            bfr[tn] = load_b_frag(&lds_w[cur][wn * 64 + tn * 16 + laneLo][0], laneHi);
#pragma unroll
        for (int tm = 0; tm < 4; ++tm) {
            const v16h a = load_a_frag(&lds_x[cur][wm * 64 + tm * 16 + laneLo][0], laneHi);
#pragma unroll
            for (int tn = 0; tn < 4; ++tn)
                c[tm][tn] = __builtin_amdgcn_wmma_f32_16x16x32_f16(
                    false, a, false, bfr[tn], (short)0, c[tm][tn], false, false);
        }
        __syncthreads();   // all waves done reading before buffer is overwritten
    }

    // ---- fused LoRA pass: one extra K=32 step with A = t tile, B[r][o] = B16[o][r] ----
#pragma unroll
    for (int i = 0; i < 4; ++i) {
        const int lin = tid + 256 * i;
        const int row = lin >> 2;
        const int kk  = (lin & 3) * 8;
        *(v8h*)&lds_x[0][row][kk] = *(const v8h*)&T[(size_t)(n0 + row) * RANK + kk];
    }
#pragma unroll
    for (int i = 0; i < 2; ++i) {
        const int lin = tid + 256 * i;
        const int row = lin >> 2;
        const int kk  = (lin & 3) * 8;
        *(v8h*)&lds_w[0][row][kk] = *(const v8h*)&B16[(size_t)(o0 + row) * RANK + kk];
    }
    __syncthreads();
    {
        v16h bfr[4];
#pragma unroll
        for (int tn = 0; tn < 4; ++tn)
            bfr[tn] = load_b_frag(&lds_w[0][wn * 64 + tn * 16 + laneLo][0], laneHi);
#pragma unroll
        for (int tm = 0; tm < 4; ++tm) {
            const v16h a = load_a_frag(&lds_x[0][wm * 64 + tm * 16 + laneLo][0], laneHi);
#pragma unroll
            for (int tn = 0; tn < 4; ++tn)
                c[tm][tn] = __builtin_amdgcn_wmma_f32_16x16x32_f16(
                    false, a, false, bfr[tn], (short)0, c[tm][tn], false, false);
        }
    }

    // ---- epilogue: bias (+relu) (+residual), store f16 or f32 ----
#pragma unroll
    for (int tm = 0; tm < 4; ++tm) {
#pragma unroll
        for (int tn = 0; tn < 4; ++tn) {
            const int colg = o0 + wn * 64 + tn * 16 + laneLo;
            const float bv = Bias[colg];
#pragma unroll
            for (int r = 0; r < 8; ++r) {
                const int rowg = n0 + wm * 64 + tm * 16 + r + 8 * laneHi;
                float v = c[tm][tn][r] + bv;
                if (RELU) v = fmaxf(v, 0.0f);
                if (ADDRES) v += Res[(size_t)rowg * DIM + colg];
                if (F16OUT) O16[(size_t)rowg * DIM + colg] = (half_t)v;
                else        O32[(size_t)rowg * DIM + colg] = v;
            }
        }
    }
}

// =====================  LoRA first GEMM: t = x @ A^T (A pre-converted f16)  =====================
__global__ __launch_bounds__(256) void lora_gemm_kernel(
    const half_t* __restrict__ X,   // [N_TOK, DIM] f16
    const half_t* __restrict__ A16, // [RANK, DIM] f16
    half_t* __restrict__ T)         // [N_TOK, RANK] f16
{
    __shared__ __align__(16) half_t lds_x[BML][KP * 2];   // 64 K-halves staged (two 32-steps)
    __shared__ __align__(16) half_t lds_w[RANK][KP * 2];

    const int tid = threadIdx.x;
    const int lane = tid & 31;
    const int wid = tid >> 5;  // 0..7, 16 rows each
    const int laneHi = lane >> 4;
    const int laneLo = lane & 15;
    const int n0 = blockIdx.x * BML;

    v8f c[2] = {v8f{}, v8f{}};

    for (int k0 = 0; k0 < DIM; k0 += 64) {
#pragma unroll
        for (int i = 0; i < 4; ++i) {
            const int lin = tid + 256 * i;
            const int row = lin >> 3;
            const int kk  = (lin & 7) * 8;
            *(v8h*)&lds_x[row][(kk & 31) + (kk >> 5) * KP] =
                *(const v8h*)&X[(size_t)(n0 + row) * DIM + k0 + kk];
        }
        {
            const int r  = tid >> 3;          // 32 rows x 64 halves = 256 x v8h
            const int kk = (tid & 7) * 8;
            *(v8h*)&lds_w[r][(kk & 31) + (kk >> 5) * KP] =
                *(const v8h*)&A16[(size_t)r * DIM + k0 + kk];
        }
        __syncthreads();

#pragma unroll
        for (int ks = 0; ks < 2; ++ks) {
            const v16h a = load_a_frag(&lds_x[wid * 16 + laneLo][ks * KP], laneHi);
#pragma unroll
            for (int tn = 0; tn < 2; ++tn) {
                const v16h b = load_b_frag(&lds_w[tn * 16 + laneLo][ks * KP], laneHi);
                c[tn] = __builtin_amdgcn_wmma_f32_16x16x32_f16(
                    false, a, false, b, (short)0, c[tn], false, false);
            }
        }
        __syncthreads();
    }

#pragma unroll
    for (int tn = 0; tn < 2; ++tn) {
        const int col = tn * 16 + laneLo;
#pragma unroll
        for (int r = 0; r < 8; ++r) {
            const int row = n0 + wid * 16 + r + 8 * laneHi;
            T[(size_t)row * RANK + col] = (half_t)c[tn][r];
        }
    }
}

// =====================  LayerNorm (one block per row)  =====================
__global__ __launch_bounds__(256) void ln_kernel(
    const float* __restrict__ Xin, const float* __restrict__ g, const float* __restrict__ b,
    half_t* __restrict__ O16, float* __restrict__ O32)
{
    const int row = blockIdx.x;
    const int tid = threadIdx.x;
    const float* xr = Xin + (size_t)row * DIM;

    float v[4];
    float s = 0.f, q = 0.f;
#pragma unroll
    for (int i = 0; i < 4; ++i) {
        v[i] = xr[tid + 256 * i];
        s += v[i];
        q += v[i] * v[i];
    }
    __shared__ float rs[256], rq[256];
    rs[tid] = s; rq[tid] = q;
    __syncthreads();
    for (int off = 128; off > 0; off >>= 1) {
        if (tid < off) { rs[tid] += rs[tid + off]; rq[tid] += rq[tid + off]; }
        __syncthreads();
    }
    const float mean = rs[0] * (1.0f / DIM);
    const float var  = rq[0] * (1.0f / DIM) - mean * mean;
    const float inv  = rsqrtf(var + 1e-5f);
#pragma unroll
    for (int i = 0; i < 4; ++i) {
        const int col = tid + 256 * i;
        const float y = (v[i] - mean) * inv * g[col] + b[col];
        O32[(size_t)row * DIM + col] = y;
        O16[(size_t)row * DIM + col] = (half_t)y;
    }
}

// =====================  elementwise converters  =====================
__global__ __launch_bounds__(256) void prep_kernel(
    const float* __restrict__ X, half_t* __restrict__ A16, float* __restrict__ R32)
{
    const size_t i = (size_t)blockIdx.x * 256 + threadIdx.x;
    const float v = X[i];
    A16[i] = (half_t)v;
    R32[i] = v;
}

__global__ __launch_bounds__(256) void cvt_f32_f16_kernel(
    const float* __restrict__ src, half_t* __restrict__ dst)
{
    const size_t i = (size_t)blockIdx.x * 256 + threadIdx.x;
    dst[i] = (half_t)src[i];
}

extern "C" void kernel_launch(void* const* d_in, const int* in_sizes, int n_in,
                              void* d_out, int out_size, void* d_ws, size_t ws_size,
                              hipStream_t stream) {
    const float* x      = (const float*)d_in[0];
    const int*   qw     = (const int*)d_in[1];
    const float* scales = (const float*)d_in[2];
    const float* bias   = (const float*)d_in[3];
    const float* lora_A = (const float*)d_in[4];
    const float* lora_B = (const float*)d_in[5];
    const float* ln_g   = (const float*)d_in[6];
    const float* ln_b   = (const float*)d_in[7];
    float* out = (float*)d_out;

    char* ws = (char*)d_ws;
    const size_t ACT16 = (size_t)N_TOK * DIM * sizeof(half_t);     // 33.5 MB
    const size_t T16SZ = (size_t)N_TOK * RANK * sizeof(half_t);    // 1 MB
    const size_t ACT32 = (size_t)N_TOK * DIM * sizeof(float);      // 67 MB
    const size_t W16SZ = (size_t)DIM * DIM * sizeof(half_t);       // 2 MB
    const size_t AB16  = (size_t)18 * RANK * DIM * sizeof(half_t); // 1.2 MB each
    half_t* act16[2];
    act16[0]      = (half_t*)(ws);
    act16[1]      = (half_t*)(ws + ACT16);
    half_t* t16   = (half_t*)(ws + 2 * ACT16);
    float*  res32 = (float*) (ws + 2 * ACT16 + T16SZ);
    float*  out32 = (float*) (ws + 2 * ACT16 + T16SZ + ACT32);
    half_t* w16   = (half_t*)(ws + 2 * ACT16 + T16SZ + 2 * ACT32);
    half_t* a16   = (half_t*)(ws + 2 * ACT16 + T16SZ + 2 * ACT32 + W16SZ);
    half_t* b16   = (half_t*)(ws + 2 * ACT16 + T16SZ + 2 * ACT32 + W16SZ + AB16);

    // one-time prep: activation f32->f16 + residual copy; lora weights f32->f16
    prep_kernel<<<(N_TOK * DIM) / 256, 256, 0, stream>>>(x, act16[0], res32);
    cvt_f32_f16_kernel<<<(18 * RANK * DIM) / 256, 256, 0, stream>>>(lora_A, a16);
    cvt_f32_f16_kernel<<<(18 * DIM * RANK) / 256, 256, 0, stream>>>(lora_B, b16);

    const dim3 gmain(DIM / BN, N_TOK / BM);
    int p = 0;
    for (int blk = 0; blk < 6; ++blk) {
        for (int j = 0; j < 3; ++j) {
            const int li = 3 * blk + j;
            const int*    Q   = qw     + (size_t)li * DIM * DIM;
            const float*  S   = scales + (size_t)li * DIM * NGRP;
            const float*  Bi  = bias   + (size_t)li * DIM;
            const half_t* A16 = a16    + (size_t)li * RANK * DIM;
            const half_t* B16 = b16    + (size_t)li * DIM * RANK;

            dequant_kernel<<<(DIM * DIM / 8) / 256, 256, 0, stream>>>(Q, S, w16);
            lora_gemm_kernel<<<N_TOK / BML, 256, 0, stream>>>(act16[p], A16, t16);

            if (j < 2) {
                qlora_gemm_kernel<1, 1, 0><<<gmain, 256, 0, stream>>>(
                    act16[p], w16, Bi, t16, B16, nullptr, act16[1 - p], nullptr);
                p ^= 1;
            } else if (blk < 5) {
                qlora_gemm_kernel<0, 0, 1><<<gmain, 256, 0, stream>>>(
                    act16[p], w16, Bi, t16, B16, res32, nullptr, out32);
                ln_kernel<<<N_TOK, 256, 0, stream>>>(
                    out32, ln_g + blk * DIM, ln_b + blk * DIM, act16[p], res32);
            } else {
                qlora_gemm_kernel<0, 0, 1><<<gmain, 256, 0, stream>>>(
                    act16[p], w16, Bi, t16, B16, res32, nullptr, out);
            }
        }
    }
}
